// FullColumn_33337536152374
// MI455X (gfx1250) — matrix operand order repaired and checked by reading, not compiled, not run on hardware
//
#include <hip/hip_runtime.h>
#include <hip/hip_bf16.h>

// ---------------------------------------------------------------------------
// Spiking conv layer for MI455X (gfx1250):
//   pot = im2col(x) @ kernel(weight)^T + 16.0   (WMMA f16 GEMM, f32 acc,
//                                                async global->LDS staging)
//   winners = sequential WTA scan over time
// ---------------------------------------------------------------------------

typedef __attribute__((ext_vector_type(16))) _Float16 v16h;
typedef __attribute__((ext_vector_type(8)))  float    v8f;

// Problem constants (from reference)
constexpr int STEP = 16, LEAK = 32, KS = 48, FODEP = 48;
constexpr int SYN = 64, NEUR = 256, OC = 10;
constexpr int BATCH = 16, TIME = 512;
constexpr int TP = TIME + KS + 1;          // 561
constexpr int GM = BATCH * TP;             // 8976   (multiple of 16)
constexpr int GMP = 9024;                  // padded to multiple of 64 (141*64)
constexpr int GN = OC * NEUR;              // 2560
constexpr int GK = SYN * KS;               // 3072
constexpr float THETA = 32.0f;
constexpr float BIASPOT = 16.0f;           // BIAS * THETA

// ---------------------------------------------------------------------------
// 1) Build flipped step/fire/leak kernel matrix, fp16: Kmat[c][s*48+dt]
// ---------------------------------------------------------------------------
__global__ void build_kmat(const float* __restrict__ weight,
                           _Float16* __restrict__ Kmat) {
  int idx = blockIdx.x * blockDim.x + threadIdx.x;
  if (idx >= GN * GK) return;
  int c = idx / GK;
  int k = idx - c * GK;
  int s = k / KS;
  int dt = k - s * KS;
  float t = (float)(KS - 1 - dt);          // flip on time axis
  float w = weight[c * SYN + s];
  float rise = t * (1.0f / STEP);
  float decay = (w * STEP - t) * (1.0f / LEAK) + w;
  float v = fmaxf(0.0f, fminf(rise, decay));
  Kmat[(size_t)c * GK + k] = (_Float16)v;
}

// ---------------------------------------------------------------------------
// 2) im2col to fp16: X[m][s*48+dt] = x_pad[b, s, t' + dt - 48],  m = b*561+t'
// ---------------------------------------------------------------------------
__global__ void im2col(const float* __restrict__ x, _Float16* __restrict__ X) {
  int idx = blockIdx.x * blockDim.x + threadIdx.x;
  constexpr int CH = GK / 8;               // 384 chunks per row
  if (idx >= GMP * CH) return;
  int m = idx / CH;
  int chunk = idx - m * CH;
  int s = chunk / 6;
  int dt0 = (chunk - s * 6) * 8;
  __align__(16) _Float16 vals[8];
  if (m < GM) {
    int b = m / TP, tp = m - b * TP;
    const float* xr = x + ((size_t)b * SYN + s) * TIME;
#pragma unroll
    for (int e = 0; e < 8; ++e) {
      int tt = tp + dt0 + e - KS;
      float v = (tt >= 0 && tt < TIME) ? xr[tt] : 0.0f;
      vals[e] = (_Float16)v;
    }
  } else {
#pragma unroll
    for (int e = 0; e < 8; ++e) vals[e] = (_Float16)0.0f;
  }
  *(uint4*)(X + (size_t)m * GK + s * KS + dt0) = *(const uint4*)vals;
}

// ---------------------------------------------------------------------------
// 3) WMMA GEMM with async global->LDS double-buffered pipeline.
//    256 threads = 8 waves; macro-tile 64(M) x 128(N); K-stage 64 (2 WMMA
//    substeps).  Wave w: m-tile = w&3, n-tiles = (w>>2)*4..+3.
//    LDS rows padded to 72 halfs (144B stride -> conflict-free b128 reads).
// ---------------------------------------------------------------------------
union FragH { v16h v; uint4 q[2]; };

static __device__ inline v8f splat8(float x) {
  v8f v;
#pragma unroll
  for (int i = 0; i < 8; ++i) v[i] = x;
  return v;
}

__global__ __launch_bounds__(256) void gemm(const _Float16* __restrict__ X,
                                            const _Float16* __restrict__ Kmat,
                                            float* __restrict__ pot) {
  constexpr int LDR = 72;                  // padded LDS row (halfs), 144B
  constexpr int NS = GK / 64;              // 48 K-stages
  __shared__ __align__(16) _Float16 As[2][64][LDR];
  __shared__ __align__(16) _Float16 Bs[2][128][LDR];

  const int tid = threadIdx.x;
  const int mBase = blockIdx.y * 64;
  const int nBase = blockIdx.x * 128;
  const int wv = tid >> 5, lane = tid & 31;
  const int mt = wv & 3, ng = wv >> 2;
  const int lr = lane & 15, lh = lane >> 4;

  // Async staging: per thread per stage, A = 2 x b128, B = 4 x b128.
  const int arow = tid >> 2, acol = (tid & 3) * 16;   // 64 rows x 64 halfs
  const int brow = tid >> 1, bcol = (tid & 1) * 32;   // 128 rows x 64 halfs
  const _Float16* aG = X + (size_t)(mBase + arow) * GK + acol;
  const _Float16* bG = Kmat + (size_t)(nBase + brow) * GK + bcol;

  // LDS byte offsets (flat->LDS uses addr[31:0]; HW adds LDS_BASE)
  const unsigned laA[2] = {
      (unsigned)(uintptr_t)&As[0][arow][acol],
      (unsigned)(uintptr_t)&As[1][arow][acol]};
  const unsigned laB[2] = {
      (unsigned)(uintptr_t)&Bs[0][brow][bcol],
      (unsigned)(uintptr_t)&Bs[1][brow][bcol]};

  // Issue one stage's async loads (global byte offset == LDS byte offset
  // delta, since INST_OFFSET applies to both addresses).
  auto issue_stage = [&](int s, int buf) {
    unsigned long long ga = (unsigned long long)(uintptr_t)(aG + (size_t)s * 64);
    unsigned long long gb = (unsigned long long)(uintptr_t)(bG + (size_t)s * 64);
    asm volatile(
        "global_load_async_to_lds_b128 %0, %1, off\n\t"
        "global_load_async_to_lds_b128 %0, %1, off offset:16"
        :: "v"(laA[buf]), "v"(ga) : "memory");
    asm volatile(
        "global_load_async_to_lds_b128 %0, %1, off\n\t"
        "global_load_async_to_lds_b128 %0, %1, off offset:16\n\t"
        "global_load_async_to_lds_b128 %0, %1, off offset:32\n\t"
        "global_load_async_to_lds_b128 %0, %1, off offset:48"
        :: "v"(laB[buf]), "v"(gb) : "memory");
  };

  v8f acc[4];
#pragma unroll
  for (int j = 0; j < 4; ++j) acc[j] = splat8(BIASPOT);   // bias folded in

  issue_stage(0, 0);

  for (int s = 0; s < NS; ++s) {
    const int buf = s & 1;
    if (s + 1 < NS) {
      issue_stage(s + 1, buf ^ 1);
      // newest 6 may stay in flight; stage s's 6 are done (in-order)
      asm volatile("s_wait_asynccnt 0x6" ::: "memory");
    } else {
      asm volatile("s_wait_asynccnt 0x0" ::: "memory");
    }
    __syncthreads();                       // everyone's stage-s data in LDS

#pragma unroll
    for (int kb = 0; kb < 64; kb += 32) {
      // A fragment (16x32 f16): lane-half h holds K = h*16 .. h*16+15
      FragH fa;
      const _Float16* ap = &As[buf][mt * 16 + lr][kb + lh * 16];
      fa.q[0] = *(const uint4*)ap;
      fa.q[1] = *(const uint4*)(ap + 8);
      // B fragments (32x16 f16): lane-half h holds K = h*16 .. h*16+15
      FragH fb[4];
#pragma unroll
      for (int j = 0; j < 4; ++j) {
        const _Float16* bp = &Bs[buf][ng * 64 + j * 16 + lr][kb + lh * 16];
        fb[j].q[0] = *(const uint4*)bp;
        fb[j].q[1] = *(const uint4*)(bp + 8);
      }
#pragma unroll
      for (int j = 0; j < 4; ++j)
        acc[j] = __builtin_amdgcn_wmma_f32_16x16x32_f16(
            false, fa.v, false, fb[j].v, (short)0, acc[j], false, false);
    }
    __syncthreads();                       // all waves done reading buf
  }

  // C/D layout: VGPR r <-> M = r + 8*h, lane low bits <-> N
#pragma unroll
  for (int j = 0; j < 4; ++j) {
    const int gn = nBase + ng * 64 + j * 16 + lr;
#pragma unroll
    for (int r = 0; r < 8; ++r) {
      const int gm = mBase + mt * 16 + lh * 8 + r;
      if (gm < GM) pot[(size_t)gm * GN + gn] = acc[j][r];
    }
  }
}

// ---------------------------------------------------------------------------
// 4) Winner-take-all time scan. One thread per (b, n); depression counter is
//    uniform over OC in the reference, so it reduces to a scalar.
// ---------------------------------------------------------------------------
__global__ void wta(const float* __restrict__ pot, float* __restrict__ out) {
  int idx = blockIdx.x * blockDim.x + threadIdx.x;
  if (idx >= BATCH * NEUR) return;
  const int b = idx >> 8, n = idx & 255;
  const float* prow = pot + (size_t)b * TP * GN + n;      // + t*GN + o*NEUR
  float* orow = out + ((size_t)b * OC * NEUR + n) * TP;   // + o*NEUR*TP + t
  int dep = 0;
  for (int t = 0; t < TP; ++t) {
    const float* pp = prow + (size_t)t * GN;
    float p[OC];
#pragma unroll
    for (int o = 0; o < OC; ++o) p[o] = pp[o * NEUR];
    int wi = 0;
    float best = p[0];
#pragma unroll
    for (int o = 1; o < OC; ++o)
      if (p[o] > best) { best = p[o]; wi = o; }           // first-max == argmax
    const float spike = (dep == 0 && best > THETA) ? 1.0f : 0.0f;
#pragma unroll
    for (int o = 0; o < OC; ++o)
      orow[(size_t)o * (NEUR * TP) + t] = (o == wi) ? spike : 0.0f;
    dep += (spike > 0.0f ? FODEP : 0) - 1;
    if (dep < 0) dep = 0;
    if (dep > FODEP - 1) dep = FODEP - 1;
  }
}

// ---------------------------------------------------------------------------
extern "C" void kernel_launch(void* const* d_in, const int* in_sizes, int n_in,
                              void* d_out, int out_size, void* d_ws, size_t ws_size,
                              hipStream_t stream) {
  const float* spikes = (const float*)d_in[0];   // [16,1,64,512] f32
  const float* weight = (const float*)d_in[1];   // [2560,64]     f32
  float* out = (float*)d_out;                    // [16,10,256,561] f32

  // Workspace layout (all 16B aligned)
  char* ws = (char*)d_ws;
  const size_t kmatBytes = (size_t)GN * GK * sizeof(_Float16);   // 15.7 MB
  const size_t xBytes    = (size_t)GMP * GK * sizeof(_Float16);  // 55.4 MB
  _Float16* Kmat = (_Float16*)ws;
  _Float16* X    = (_Float16*)(ws + kmatBytes);
  float*    pot  = (float*)(ws + kmatBytes + xBytes);            // 91.9 MB

  {
    int total = GN * GK;
    build_kmat<<<(total + 255) / 256, 256, 0, stream>>>(weight, Kmat);
  }
  {
    int total = GMP * (GK / 8);
    im2col<<<(total + 255) / 256, 256, 0, stream>>>(spikes, X);
  }
  {
    dim3 grid(GN / 128, GMP / 64);   // 20 x 141
    gemm<<<grid, 256, 0, stream>>>(X, Kmat, pot);
  }
  {
    wta<<<(BATCH * NEUR + 255) / 256, 256, 0, stream>>>(pot, out);
  }
}
